// Layer_69604239999638
// MI455X (gfx1250) — compile-verified, matrix operand order checked by reading, not run on hardware
//
#include <hip/hip_runtime.h>

typedef __bf16 bf16;
typedef __attribute__((ext_vector_type(16))) __bf16 v16bf;
typedef __attribute__((ext_vector_type(8)))  float  v8f;

#define B_  16
#define S_  512
#define D_  1024
#define H_  16
#define HD_ 64
#define FF_ 4096
#define M_  8192   // B_*S_

// ---------------------------------------------------------------- WMMA helpers

__device__ __forceinline__ v8f wmma_bf16(v16bf a, v16bf b, v8f c) {
  // D = A(16x32) * B(32x16) + C, fp32 accumulate
  return __builtin_amdgcn_wmma_f32_16x16x32_bf16(
      /*neg_a=*/false, a, /*neg_b=*/false, b,
      /*c_mod=*/(short)0, c, /*reuse_a=*/false, /*reuse_b=*/false);
}

union FragU { v16bf v; uint4 u[2]; };

// Load one 16-bit WMMA operand fragment. `rowp` points at (row, k=0) of a
// row-major slab whose row is this lane's M (for A) or N (for B^T layout).
// Per ISA layout: elem i -> K = (i<8?0:16) + (lane>=16?8:0) + i%8, i.e. two
// contiguous 8-element (16B) runs at +k0 and +16+k0.
__device__ __forceinline__ v16bf frag_ld(const bf16* __restrict__ rowp) {
  const int lane = threadIdx.x & 31;
  const int k0 = (lane >> 4) << 3;   // 0 or 8
  FragU f;
  f.u[0] = *(const uint4*)(rowp + k0);
  f.u[1] = *(const uint4*)(rowp + 16 + k0);
  return f.v;
}

__device__ __forceinline__ float gelu_f(float x) {
  const float t = tanhf(0.7978845608028654f * (x + 0.044715f * x * x * x));
  return 0.5f * x * (1.f + t);
}

// --------------------------------------------------- Tensor Data Mover staging
#if __has_builtin(__builtin_amdgcn_tensor_load_to_lds) && \
    __has_builtin(__builtin_amdgcn_s_wait_tensorcnt)
#define USE_TDM 1
typedef __attribute__((ext_vector_type(4))) unsigned int t_v4u;
typedef __attribute__((ext_vector_type(8))) int t_v8i;
typedef __attribute__((ext_vector_type(4))) int t_v4i;

// DMA one 128x32 bf16 tile (row-major, row stride lda_elems) into LDS at
// byte offset lds_byte_off. D# per CDNA5 ISA ch.8:
//  g0: count=1 | lds_addr | global_addr[56:0] | type=2
//  g1: data_size=1(2B); tensor_dim0/1 = 1<<30 (never OOB); tile=32x128;
//      tensor_dim0_stride = lda_elems. remaining groups = 0 (2-D tensor).
__device__ __forceinline__ void tdm_tile_128x32(const bf16* g, int lda_elems,
                                                unsigned lds_byte_off) {
  const unsigned long long ga = (unsigned long long)g;
  const t_v4u g0 = { 1u,
                     lds_byte_off,
                     (unsigned)(ga & 0xFFFFFFFFull),
                     (unsigned)((ga >> 32) & 0x01FFFFFFull) | (2u << 30) };
  const t_v8i g1 = { (int)(1u << 16),               // data_size=2B, mask=0
                     0,                             // abar=0, dim0.lo16=0
                     0x4000,                        // dim0.hi16 (1<<30), dim1.lo16=0
                     (int)((32u << 16) | 0x4000u),  // dim1.hi16 (1<<30), tile_dim0=32
                     128,                           // tile_dim1=128, tile_dim2=0
                     lda_elems,                     // dim0_stride[31:0]
                     0, 0 };
  const t_v4i gz4 = { 0, 0, 0, 0 };
  const t_v8i gz8 = { 0, 0, 0, 0, 0, 0, 0, 0 };
  // 6-arg form (upstream clang-23 / therock-10.0 headers)
  __builtin_amdgcn_tensor_load_to_lds(g0, g1, gz4, gz4, gz8, 0);
}
#endif

// ------------------------------------------------- weight convert + transpose
// w: K x N fp32 (row-major)  ->  wt: N x K bf16 (row-major), i.e. B^T layout.
__global__ __launch_bounds__(256) void wcvt(const float* __restrict__ w,
                                            bf16* __restrict__ wt,
                                            int K, int N) {
  const size_t idx = (size_t)blockIdx.x * 256 + threadIdx.x;
  const int k = (int)(idx / (size_t)N);
  const int n = (int)(idx % (size_t)N);
  wt[(size_t)n * K + k] = (bf16)w[idx];
}

// ------------------------------------------------------------------ LayerNorm
__global__ __launch_bounds__(256) void ln_k(const float* __restrict__ x,
                                            const float* __restrict__ scale,
                                            bf16* __restrict__ out) {
  __shared__ float red[256];
  const int row = blockIdx.x, t = threadIdx.x;
  const float4 v = *(const float4*)(x + (size_t)row * D_ + (t << 2));
  red[t] = v.x + v.y + v.z + v.w;
  __syncthreads();
  for (int o = 128; o > 0; o >>= 1) { if (t < o) red[t] += red[t + o]; __syncthreads(); }
  const float mean = red[0] * (1.f / D_);
  __syncthreads();
  const float d0 = v.x - mean, d1 = v.y - mean, d2 = v.z - mean, d3 = v.w - mean;
  red[t] = d0 * d0 + d1 * d1 + d2 * d2 + d3 * d3;
  __syncthreads();
  for (int o = 128; o > 0; o >>= 1) { if (t < o) red[t] += red[t + o]; __syncthreads(); }
  const float inv = rsqrtf(red[0] * (1.f / D_) + 1e-6f);
  const float* s4 = scale + (t << 2);
  bf16* op = out + (size_t)row * D_ + (t << 2);
  op[0] = (bf16)(d0 * inv * s4[0]);
  op[1] = (bf16)(d1 * inv * s4[1]);
  op[2] = (bf16)(d2 * inv * s4[2]);
  op[3] = (bf16)(d3 * inv * s4[3]);
}

// ----------------------------------------------------------------- tiled GEMM
// C(MxN) = A(MxK, row-major, lda) * Bt^T  where Bt is NxK row-major (ldb).
// Block tile 128x128x32, 8 waves; each wave: 2x4 grid of 16x16 WMMA tiles.
// A/B tiles staged into LDS by the Tensor Data Mover (wave 0 issues the DMA,
// TENSORcnt + barrier publishes), double-buffered.
// MODE 0: store bf16 C               MODE 1: store bf16 V^T per-head scatter
// MODE 2: Cf = resid + alpha*acc + bias (fp32)
// MODE 3: Cb = gelu(acc + bias) (bf16)
template <int MODE>
__global__ __launch_bounds__(256) void gemm_bt(
    const bf16* __restrict__ A, int lda,
    const bf16* __restrict__ Bt, int ldb,
    int N, int K,
    bf16* __restrict__ Cb, float* __restrict__ Cf,
    const float* __restrict__ resid, const float* __restrict__ bias,
    float alpha) {
  __shared__ bf16 As[2][128][32];
  __shared__ bf16 Bs[2][128][32];
  const int tid = threadIdx.x;
  const int lane = tid & 31, wid = tid >> 5;
  const int m0 = blockIdx.y * 128, n0 = blockIdx.x * 128;
  const int wm = (wid & 3) << 5;   // wave row offset: 0/32/64/96
  const int wn = (wid >> 2) << 6;  // wave col offset: 0/64

  v8f acc[2][4] = {};
  const int KT = K >> 5;

#ifdef USE_TDM
  const unsigned ldsA = (unsigned)(unsigned long long)&As[0][0][0];
  const unsigned ldsB = (unsigned)(unsigned long long)&Bs[0][0][0];
  const bf16* Abase = A + (size_t)m0 * lda;
  const bf16* Bbase = Bt + (size_t)n0 * ldb;
  if (wid == 0) {
    tdm_tile_128x32(Abase, lda, ldsA);
    tdm_tile_128x32(Bbase, ldb, ldsB);
  }
#else
  const int srow = tid >> 1;        // staging: 2 threads per 32-elem row
  const int scol = (tid & 1) << 4;  // 0 or 16
  auto stage = [&](int buf, int kt) {
    const bf16* ag = A + (size_t)(m0 + srow) * lda + (kt << 5) + scol;
    *(uint4*)&As[buf][srow][scol]     = *(const uint4*)ag;
    *(uint4*)&As[buf][srow][scol + 8] = *(const uint4*)(ag + 8);
    const bf16* bg = Bt + (size_t)(n0 + srow) * ldb + (kt << 5) + scol;
    *(uint4*)&Bs[buf][srow][scol]     = *(const uint4*)bg;
    *(uint4*)&Bs[buf][srow][scol + 8] = *(const uint4*)(bg + 8);
  };
  stage(0, 0);
  __syncthreads();
#endif

  for (int kt = 0; kt < KT; ++kt) {
    const int buf = kt & 1;
#ifdef USE_TDM
    if (wid == 0) {
      if (kt + 1 < KT) {
        tdm_tile_128x32(Abase + ((size_t)(kt + 1) << 5), lda,
                        ldsA + (unsigned)((buf ^ 1) * 8192));
        tdm_tile_128x32(Bbase + ((size_t)(kt + 1) << 5), ldb,
                        ldsB + (unsigned)((buf ^ 1) * 8192));
        __builtin_amdgcn_s_wait_tensorcnt(2);  // oldest pair (this buf) done
      } else {
        __builtin_amdgcn_s_wait_tensorcnt(0);
      }
    }
    __syncthreads();   // publish buf to all waves
#else
    if (kt + 1 < KT) stage(buf ^ 1, kt + 1);
#endif
    v16bf af[2], bfr[4];
#pragma unroll
    for (int tm = 0; tm < 2; ++tm)
      af[tm] = frag_ld(&As[buf][wm + (tm << 4) + (lane & 15)][0]);
#pragma unroll
    for (int tn = 0; tn < 4; ++tn)
      bfr[tn] = frag_ld(&Bs[buf][wn + (tn << 4) + (lane & 15)][0]);
#pragma unroll
    for (int tm = 0; tm < 2; ++tm)
#pragma unroll
      for (int tn = 0; tn < 4; ++tn)
        acc[tm][tn] = wmma_bf16(af[tm], bfr[tn], acc[tm][tn]);
    __syncthreads();   // readers done before buf is overwritten
  }

  const int rbase = (lane >> 4) << 3;  // C layout: lanes 16-31 hold M+8
  const int cl = lane & 15;            // N = lane%16
#pragma unroll
  for (int tm = 0; tm < 2; ++tm)
#pragma unroll
    for (int tn = 0; tn < 4; ++tn)
#pragma unroll
      for (int v = 0; v < 8; ++v) {
        const int m = m0 + wm + (tm << 4) + rbase + v;
        const int n = n0 + wn + (tn << 4) + cl;
        const float val = acc[tm][tn][v];
        if constexpr (MODE == 0) {
          Cb[(size_t)m * N + n] = (bf16)val;
        } else if constexpr (MODE == 1) {
          const int b = m >> 9, i = m & (S_ - 1), h = n >> 6, d = n & (HD_ - 1);
          Cb[(((size_t)b * H_ + h) * HD_ + d) * S_ + i] = (bf16)val;
        } else if constexpr (MODE == 2) {
          const float bb = bias ? bias[n] : 0.f;
          Cf[(size_t)m * N + n] = resid[(size_t)m * N + n] + alpha * val + bb;
        } else {
          Cb[(size_t)m * N + n] = (bf16)gelu_f(val + bias[n]);
        }
      }
}

// ------------------------------------------------------------------ attention
// grid: B_*H_*(S_/16) blocks; each block: one (b,h), 16 query rows.
__global__ __launch_bounds__(256) void attn_k(const bf16* __restrict__ q,
                                              const bf16* __restrict__ k,
                                              const bf16* __restrict__ vt,
                                              bf16* __restrict__ attn) {
  __shared__ float sc[16][512];   // 32 KB (reused as 16x64 out accumulator)
  __shared__ bf16  p[16][512];    // 16 KB
  __shared__ float red[16][16];
  __shared__ float rstat[16];

  const int blk = blockIdx.x;
  const int qt = blk & 31;
  const int h  = (blk >> 5) & (H_ - 1);
  const int b  = blk >> 9;
  const int i0 = qt << 4;
  const int tid = threadIdx.x, wid = tid >> 5, lane = tid & 31;
  const int cl = lane & 15, rbase = (lane >> 4) << 3;

  // ---- scores: wave `wid` computes columns [wid*64, wid*64+64)
  {
    v8f acc[4] = {};
    const bf16* qrow = q + ((size_t)(b * S_ + i0 + cl) * D_ + h * HD_);
#pragma unroll
    for (int kt = 0; kt < 2; ++kt) {
      const v16bf aq = frag_ld(qrow + (kt << 5));
#pragma unroll
      for (int tn = 0; tn < 4; ++tn) {
        const bf16* krow =
            k + ((size_t)(b * S_ + (wid << 6) + (tn << 4) + cl) * D_ + h * HD_ + (kt << 5));
        acc[tn] = wmma_bf16(aq, frag_ld(krow), acc[tn]);
      }
    }
#pragma unroll
    for (int tn = 0; tn < 4; ++tn)
#pragma unroll
      for (int v = 0; v < 8; ++v)
        sc[rbase + v][(wid << 6) + (tn << 4) + cl] = acc[tn][v] * 0.125f;
  }
  __syncthreads();

  // ---- softmax: 16 rows x 16 partitions of 32 columns
  {
    const int row = tid >> 4, part = tid & 15;
    float m = -3.4e38f;
    for (int j = part * 32; j < part * 32 + 32; ++j) m = fmaxf(m, sc[row][j]);
    red[row][part] = m;
    __syncthreads();
    if (part == 0) {
      float mm = red[row][0];
      for (int i2 = 1; i2 < 16; ++i2) mm = fmaxf(mm, red[row][i2]);
      rstat[row] = mm;
    }
    __syncthreads();
    m = rstat[row];
    float s = 0.f;
    for (int j = part * 32; j < part * 32 + 32; ++j) {
      const float e = __expf(sc[row][j] - m);
      sc[row][j] = e;
      s += e;
    }
    red[row][part] = s;
    __syncthreads();
    if (part == 0) {
      float ss = 0.f;
      for (int i2 = 0; i2 < 16; ++i2) ss += red[row][i2];
      rstat[row] = 1.f / ss;
    }
    __syncthreads();
    const float inv = rstat[row];
    for (int j = part * 32; j < part * 32 + 32; ++j)
      p[row][j] = (bf16)(sc[row][j] * inv);
  }
  __syncthreads();

  // ---- zero output accumulator (overlay on sc)
  float* oacc = &sc[0][0];  // 16*64 floats
  for (int idx = tid; idx < 16 * 64; idx += 256) oacc[idx] = 0.f;
  __syncthreads();

  // ---- PV: wave `wid` handles j in [wid*64, wid*64+64), split-K reduce
  {
    v8f acc[4] = {};
#pragma unroll
    for (int kt = 0; kt < 2; ++kt) {
      const v16bf ap = frag_ld(&p[cl][(wid << 6) + (kt << 5)]);
#pragma unroll
      for (int tn = 0; tn < 4; ++tn) {
        const bf16* vrow = vt + (((size_t)(b * H_ + h) * HD_ + (tn << 4) + cl) * S_ +
                                 (wid << 6) + (kt << 5));
        acc[tn] = wmma_bf16(ap, frag_ld(vrow), acc[tn]);
      }
    }
#pragma unroll
    for (int tn = 0; tn < 4; ++tn)
#pragma unroll
      for (int v = 0; v < 8; ++v)
        atomicAdd(&oacc[(rbase + v) * 64 + (tn << 4) + cl], acc[tn][v]);
  }
  __syncthreads();

  for (int idx = tid; idx < 16 * 64; idx += 256) {
    const int r = idx >> 6, d = idx & 63;
    attn[(size_t)(b * S_ + i0 + r) * D_ + h * HD_ + d] = (bf16)oacc[idx];
  }
}

// --------------------------------------------------------------------- driver
extern "C" void kernel_launch(void* const* d_in, const int* in_sizes, int n_in,
                              void* d_out, int out_size, void* d_ws, size_t ws_size,
                              hipStream_t stream) {
  (void)in_sizes; (void)n_in; (void)out_size; (void)ws_size;
  const float* x   = (const float*)d_in[0];
  const float* wq  = (const float*)d_in[1];
  const float* wk  = (const float*)d_in[2];
  const float* wv  = (const float*)d_in[3];
  const float* wo  = (const float*)d_in[4];
  const float* ln1 = (const float*)d_in[5];
  const float* ln2 = (const float*)d_in[6];
  const float* w1  = (const float*)d_in[7];
  const float* b1  = (const float*)d_in[8];
  const float* w2  = (const float*)d_in[9];
  const float* b2  = (const float*)d_in[10];

  char* wp = (char*)d_ws;
  auto take = [&](size_t bytes) {
    char* r = wp;
    wp += (bytes + 255) & ~(size_t)255;
    return r;
  };
  bf16* wqt = (bf16*)take((size_t)D_ * D_ * 2);
  bf16* wkt = (bf16*)take((size_t)D_ * D_ * 2);
  bf16* wvt = (bf16*)take((size_t)D_ * D_ * 2);
  bf16* wot = (bf16*)take((size_t)D_ * D_ * 2);
  bf16* w1t = (bf16*)take((size_t)D_ * FF_ * 2);
  bf16* w2t = (bf16*)take((size_t)D_ * FF_ * 2);
  bf16* c   = (bf16*)take((size_t)M_ * D_ * 2);
  bf16* qb  = (bf16*)take((size_t)M_ * D_ * 2);
  bf16* kb  = (bf16*)take((size_t)M_ * D_ * 2);
  bf16* vtb = (bf16*)take((size_t)M_ * D_ * 2);
  bf16* ab  = (bf16*)take((size_t)M_ * D_ * 2);
  bf16* h1  = (bf16*)take((size_t)M_ * FF_ * 2);
  float* xa = (float*)take((size_t)M_ * D_ * 4);

  const dim3 blk(256);
  wcvt<<<(D_ * D_) / 256, blk, 0, stream>>>(wq, wqt, D_, D_);
  wcvt<<<(D_ * D_) / 256, blk, 0, stream>>>(wk, wkt, D_, D_);
  wcvt<<<(D_ * D_) / 256, blk, 0, stream>>>(wv, wvt, D_, D_);
  wcvt<<<(D_ * D_) / 256, blk, 0, stream>>>(wo, wot, D_, D_);
  wcvt<<<(D_ * FF_) / 256, blk, 0, stream>>>(w1, w1t, D_, FF_);
  wcvt<<<(FF_ * D_) / 256, blk, 0, stream>>>(w2, w2t, FF_, D_);

  const dim3 gD(D_ / 128, M_ / 128);
  const dim3 gF(FF_ / 128, M_ / 128);

  ln_k<<<M_, blk, 0, stream>>>(x, ln1, c);
  gemm_bt<0><<<gD, blk, 0, stream>>>(c, D_, wqt, D_, D_, D_, qb, nullptr, nullptr, nullptr, 0.f);
  gemm_bt<0><<<gD, blk, 0, stream>>>(c, D_, wkt, D_, D_, D_, kb, nullptr, nullptr, nullptr, 0.f);
  gemm_bt<1><<<gD, blk, 0, stream>>>(c, D_, wvt, D_, D_, D_, vtb, nullptr, nullptr, nullptr, 0.f);
  attn_k<<<B_ * H_ * (S_ / 16), blk, 0, stream>>>(qb, kb, vtb, ab);
  gemm_bt<2><<<gD, blk, 0, stream>>>(ab, D_, wot, D_, D_, D_, nullptr, xa, x, nullptr, 2.f);
  ln_k<<<M_, blk, 0, stream>>>(xa, ln2, c);
  gemm_bt<3><<<gF, blk, 0, stream>>>(c, D_, w1t, D_, FF_, D_, h1, nullptr, nullptr, b1, 0.f);
  gemm_bt<2><<<gD, blk, 0, stream>>>(h1, FF_, w2t, FF_, D_, FF_, nullptr, xa, xa, b2, 1.f);
  ln_k<<<M_, blk, 0, stream>>>(xa, ln2, c);
  gemm_bt<3><<<gF, blk, 0, stream>>>(c, D_, w1t, D_, FF_, D_, h1, nullptr, nullptr, b1, 0.f);
  gemm_bt<2><<<gD, blk, 0, stream>>>(h1, FF_, w2t, FF_, D_, FF_, nullptr, (float*)d_out, xa, b2, 1.f);
}
